// STCHGAT_76124000354330
// MI455X (gfx1250) — compile-verified, matrix operand order checked by reading, not run on hardware
//
#include <hip/hip_runtime.h>
#include <hip/hip_bf16.h>
#include <math.h>

typedef __attribute__((ext_vector_type(16))) _Float16 v16h;
typedef __attribute__((ext_vector_type(8)))  float    v8f;
typedef __attribute__((ext_vector_type(4)))  float    vf4;

#define FLAG_ACC  1
#define FLAG_RELU 2

// Model dims
#define BB 16
#define NN 100
#define TT 96
#define FF 26
#define HH 128
#define NHH 4
#define EE 16
#define BN (BB*NN)          // 1600
#define PP (BN*TT)          // 153600

__device__ __forceinline__ float sigmoidf_(float x) { return 1.0f / (1.0f + __expf(-x)); }

// Load a 16-half WMMA fragment from contiguous-K f32 data.
// p must already include the row base + hi*8 lane offset; k0 is the K-chunk base.
// Layout: halves 0..7 <- p[k0..k0+7], halves 8..15 <- p[k0+16..k0+23].
__device__ __forceinline__ v16h load_frag_fast(const float* __restrict__ p, int k0)
{
    const vf4 a = *(const vf4*)(p + k0);
    const vf4 b = *(const vf4*)(p + k0 + 4);
    const vf4 c = *(const vf4*)(p + k0 + 16);
    const vf4 d = *(const vf4*)(p + k0 + 20);
    v16h r;
#pragma unroll
    for (int i = 0; i < 4; ++i) {
        r[i]      = (_Float16)a[i];
        r[4 + i]  = (_Float16)b[i];
        r[8 + i]  = (_Float16)c[i];
        r[12 + i] = (_Float16)d[i];
    }
    return r;
}

// ---------------------------------------------------------------------------
// Fast GEMM, K fixed at 128 (all hot GEMMs in this model have K=128):
// C(m,n) = sum_k A(m,k)*B(n-major,k) [+bias] ; M%16==0, N%32==0, contiguous K.
// One wave -> 16x32 tile (one A fragment, two B fragments, two accumulators).
// FLAGS/HASBIAS are compile-time: branch-free epilogue, fully unrolled K loop.
// ---------------------------------------------------------------------------
template<int FLAGS, bool HASBIAS>
__global__ __launch_bounds__(32) void gemm_wmma_fast(
    const float* __restrict__ A, const float* __restrict__ B,
    const float* __restrict__ bias, float* __restrict__ C,
    int msA, int nsB, int ldc)
{
    const int lane = threadIdx.x & 31;
    const int lo = lane & 15, hi = lane >> 4;
    const int m0 = blockIdx.x * 16, n0 = blockIdx.y * 32;

    const float* ap  = A + (long long)(m0 + lo) * msA + hi * 8;
    const float* bp0 = B + (long long)(n0 + lo) * nsB + hi * 8;
    const float* bp1 = B + (long long)(n0 + 16 + lo) * nsB + hi * 8;

    v8f acc0 = {0.f, 0.f, 0.f, 0.f, 0.f, 0.f, 0.f, 0.f};
    v8f acc1 = {0.f, 0.f, 0.f, 0.f, 0.f, 0.f, 0.f, 0.f};
#pragma unroll
    for (int k0 = 0; k0 < 128; k0 += 32) {
        const v16h af = load_frag_fast(ap, k0);
        const v16h b0 = load_frag_fast(bp0, k0);
        const v16h b1 = load_frag_fast(bp1, k0);
        acc0 = __builtin_amdgcn_wmma_f32_16x16x32_f16(false, af, false, b0,
                                                      (short)0, acc0, false, false);
        acc1 = __builtin_amdgcn_wmma_f32_16x16x32_f16(false, af, false, b1,
                                                      (short)0, acc1, false, false);
    }

    const int nc0 = n0 + lo, nc1 = n0 + 16 + lo;
    const float bad0 = HASBIAS ? bias[nc0] : 0.f;
    const float bad1 = HASBIAS ? bias[nc1] : 0.f;
#pragma unroll
    for (int r = 0; r < 8; ++r) {
        const long long row = m0 + (hi << 3) + r;
        float v0 = acc0[r] + bad0;
        float v1 = acc1[r] + bad1;
        const long long o0 = row * ldc + nc0;
        const long long o1 = row * ldc + nc1;
        if (FLAGS & FLAG_ACC)  { v0 += C[o0]; v1 += C[o1]; }
        if (FLAGS & FLAG_RELU) { v0 = fmaxf(v0, 0.f); v1 = fmaxf(v1, 0.f); }
        C[o0] = v0;
        C[o1] = v1;
    }
}

// ---------------------------------------------------------------------------
// Generic batched WMMA GEMM (irregular shapes):
//   C[b](m,n) = sum_k A[b](m,k)*B[b](k,n) [+bias(n)]
// A element (m,k) at A + b*sA + m*msA + k*ksA
// B element (k,n) at B + b*sB + n*nsB + k*ksB
// Guards use clamped unconditional loads + selects (no exec-mask branches).
// ---------------------------------------------------------------------------
__global__ __launch_bounds__(32) void gemm_wmma(
    const float* __restrict__ A, const float* __restrict__ B,
    const float* __restrict__ bias, float* __restrict__ C,
    int M, int N, int K,
    int msA, int ksA, int nsB, int ksB, int ldc,
    long long sA, long long sB, long long sC, int flags)
{
    const int lane = threadIdx.x & 31;
    const int lo = lane & 15, hi = lane >> 4;
    const int m0 = blockIdx.x * 16, n0 = blockIdx.y * 16;
    const int bz = blockIdx.z;
    const float* Ab = A + (long long)bz * sA;
    const float* Bb = B + (long long)bz * sB;
    float*       Cb = C + (long long)bz * sC;

    const int arow = m0 + lo;
    const bool arow_ok = (arow < M);
    const int arow_c = arow_ok ? arow : (M - 1);
    const int ncol = n0 + lo;

    v8f acc = {0.f, 0.f, 0.f, 0.f, 0.f, 0.f, 0.f, 0.f};
    for (int k0 = 0; k0 < K; k0 += 32) {
        v16h af, bf;
#pragma unroll
        for (int j = 0; j < 16; ++j) {
            const int kk = k0 + ((j >> 3) << 4) + (hi << 3) + (j & 7);
            const bool kok = (kk < K);
            const int kc = kok ? kk : (K - 1);
            // Loads are unconditional (clamped in-bounds); zeroing via select.
            const float av = Ab[(long long)arow_c * msA + (long long)kc * ksA];
            const float bv = Bb[(long long)ncol * nsB + (long long)kc * ksB];
            af[j] = (_Float16)((arow_ok && kok) ? av : 0.f);
            bf[j] = (_Float16)(kok ? bv : 0.f);
        }
        acc = __builtin_amdgcn_wmma_f32_16x16x32_f16(false, af, false, bf,
                                                     (short)0, acc, false, false);
    }

    const float badd = bias ? bias[ncol] : 0.f;
#pragma unroll
    for (int r = 0; r < 8; ++r) {
        const int row = m0 + (hi << 3) + r;
        if (row < M) {
            const long long off = (long long)row * ldc + ncol;
            float v = acc[r] + badd;
            if (flags & FLAG_ACC)  v += Cb[off];
            if (flags & FLAG_RELU) v = fmaxf(v, 0.f);
            Cb[off] = v;
        }
    }
}

// ---------------------------------------------------------------------------
// Fused attention per (bn, head): S = softmax(Q K^T / sqrt(32)); O = S V
// Scores kept entirely in LDS (96 x 104 f32, 16B-aligned rows). 8 waves.
// ---------------------------------------------------------------------------
#define SPITCH 104
__global__ __launch_bounds__(256) void attention_fused(
    const float* __restrict__ qkv, float* __restrict__ ao)
{
    __shared__ float S[TT][SPITCH];
    const int bh = blockIdx.x;
    const int bn = bh >> 2, head = bh & 3;
    const float* base = qkv + (long long)bn * TT * 384;
    const float* qp = base + head * 32;
    const float* kp = base + 128 + head * 32;
    const float* vp = base + 256 + head * 32;
    const int wave = threadIdx.x >> 5;
    const int lane = threadIdx.x & 31;
    const int lo = lane & 15, hi = lane >> 4;

    // Phase 1: scores (6x6 tiles of 16x16, K = dh = 32 -> one WMMA each)
    for (int tile = wave; tile < 36; tile += 8) {
        const int tm = tile / 6, tn = tile % 6;
        const v16h af = load_frag_fast(qp + (tm * 16 + lo) * 384 + hi * 8, 0);
        const v16h bf = load_frag_fast(kp + (tn * 16 + lo) * 384 + hi * 8, 0);
        v8f acc = {0.f, 0.f, 0.f, 0.f, 0.f, 0.f, 0.f, 0.f};
        acc = __builtin_amdgcn_wmma_f32_16x16x32_f16(false, af, false, bf,
                                                     (short)0, acc, false, false);
#pragma unroll
        for (int r = 0; r < 8; ++r)
            S[tm * 16 + (hi << 3) + r][tn * 16 + lo] = acc[r] * 0.17677669529663687f;
    }
    __syncthreads();

    // Phase 2: row softmax (threads 0..95 own one row each)
    if (threadIdx.x < TT) {
        const int row = threadIdx.x;
        float mx = -3.0e38f;
        for (int j = 0; j < TT; ++j) mx = fmaxf(mx, S[row][j]);
        float sum = 0.f;
        for (int j = 0; j < TT; ++j) { float e = __expf(S[row][j] - mx); S[row][j] = e; sum += e; }
        const float inv = 1.f / sum;
        for (int j = 0; j < TT; ++j) S[row][j] *= inv;
    }
    __syncthreads();

    // Phase 3: O = S @ V (6x2 tiles, K = 96 -> 3 WMMAs each)
    for (int tile = wave; tile < 12; tile += 8) {
        const int tm = tile >> 1, tn = tile & 1;
        const float* srow = &S[tm * 16 + lo][0] + hi * 8;
        v8f acc = {0.f, 0.f, 0.f, 0.f, 0.f, 0.f, 0.f, 0.f};
        for (int k0 = 0; k0 < TT; k0 += 32) {
            const v16h af = load_frag_fast(srow, k0);   // ds_load_b128 path
            v16h bf;
#pragma unroll
            for (int j = 0; j < 16; ++j) {
                const int kk = k0 + ((j >> 3) << 4) + (hi << 3) + (j & 7);
                bf[j] = (_Float16)vp[kk * 384 + tn * 16 + lo];
            }
            acc = __builtin_amdgcn_wmma_f32_16x16x32_f16(false, af, false, bf,
                                                         (short)0, acc, false, false);
        }
#pragma unroll
        for (int r = 0; r < 8; ++r) {
            const int row = tm * 16 + (hi << 3) + r;
            ao[((long long)bn * TT + row) * HH + head * 32 + tn * 16 + lo] = acc[r];
        }
    }
}

// ---------------------------------------------------------------------------
// LayerNorm over 128 cols, one wave per row. Optional residual add + ReLU.
// ---------------------------------------------------------------------------
__global__ __launch_bounds__(256) void layernorm_kernel(
    const float* __restrict__ in, const float* __restrict__ resid,
    const float* __restrict__ g, const float* __restrict__ b,
    float* __restrict__ out, int rows, int relu)
{
    const int wave = threadIdx.x >> 5;
    const int lane = threadIdx.x & 31;
    const int row = blockIdx.x * 8 + wave;
    if (row >= rows) return;
    const long long boff = (long long)row * HH;
    float x[4];
    float s = 0.f, sq = 0.f;
#pragma unroll
    for (int i = 0; i < 4; ++i) {
        const int c = lane + 32 * i;
        float v = in[boff + c];
        if (resid) v += resid[boff + c];
        x[i] = v; s += v; sq += v * v;
    }
#pragma unroll
    for (int off = 16; off > 0; off >>= 1) {
        s  += __shfl_xor(s,  off, 32);
        sq += __shfl_xor(sq, off, 32);
    }
    const float mean = s * (1.f / 128.f);
    const float var  = sq * (1.f / 128.f) - mean * mean;
    const float rstd = rsqrtf(var + 1e-5f);
#pragma unroll
    for (int i = 0; i < 4; ++i) {
        const int c = lane + 32 * i;
        float v = (x[i] - mean) * rstd * g[c] + b[c];
        if (relu) v = fmaxf(v, 0.f);
        out[boff + c] = v;
    }
}

// LSTM cell: gates (rows x 512, order i,f,g,o) -> h,c
__global__ __launch_bounds__(256) void lstm_cell(
    const float* __restrict__ gates, const float* __restrict__ cprev,
    float* __restrict__ hout, float* __restrict__ cout, int nrows, int first)
{
    const int idx = blockIdx.x * blockDim.x + threadIdx.x;
    if (idx >= nrows * HH) return;
    const int row = idx >> 7, col = idx & 127;
    const float* gr = gates + (long long)row * 512;
    const float gi = gr[col], gf = gr[128 + col], gg = gr[256 + col], go = gr[384 + col];
    const float cp = first ? 0.f : cprev[idx];
    const float c = sigmoidf_(gf) * cp + sigmoidf_(gi) * tanhf(gg);
    const float h = sigmoidf_(go) * tanhf(c);
    cout[idx] = c;
    hout[idx] = h;
}

__global__ void add_vec(const float* a, const float* b, float* o, int n)
{
    int i = blockIdx.x * blockDim.x + threadIdx.x;
    if (i < n) o[i] = a[i] + b[i];
}

__global__ void edge_deg(const float* __restrict__ Hinc, float* __restrict__ deg)
{
    int e = threadIdx.x;
    if (e < EE) {
        float s = 0.f;
        for (int n = 0; n < NN; ++n) s += Hinc[n * EE + e];
        deg[e] = s;
    }
}

__global__ void scale_edges(float* __restrict__ ee, const float* __restrict__ deg, int n)
{
    int i = blockIdx.x * blockDim.x + threadIdx.x;
    if (i < n) {
        int e = (i >> 7) & (EE - 1);
        ee[i] = ee[i] / (deg[e] + 1e-8f);
    }
}

__global__ void edge_score(const float* __restrict__ ee, const float* __restrict__ Wa,
                           const float* __restrict__ ba, float* __restrict__ sc)
{
    int i = blockIdx.x * blockDim.x + threadIdx.x;
    if (i < BB * EE) {
        float s = 0.f;
        for (int c = 0; c < HH; ++c) s += ee[(long long)i * HH + c] * Wa[c];
        sc[i] = s + ba[0];
    }
}

__global__ void edge_softmax(const float* __restrict__ sc, const float* __restrict__ Hinc,
                             float* __restrict__ w)
{
    int i = blockIdx.x * blockDim.x + threadIdx.x;
    if (i >= BB * NN) return;
    const int b = i / NN, n = i % NN;
    float vals[EE];
    bool any = false;
    float mx = -3.0e38f;
    for (int e = 0; e < EE; ++e) {
        const bool m = Hinc[n * EE + e] > 0.f;
        const float v = m ? sc[b * EE + e] : -1e9f;
        vals[e] = v;
        if (m) any = true;
        mx = fmaxf(mx, v);
    }
    float sum = 0.f;
    for (int e = 0; e < EE; ++e) { vals[e] = __expf(vals[e] - mx); sum += vals[e]; }
    const float inv = 1.f / sum;
    for (int e = 0; e < EE; ++e)
        w[(long long)i * EE + e] = any ? vals[e] * inv : 0.f;
}

__global__ void spatial_combine(const float* __restrict__ hb2, const float* __restrict__ hg,
                                const float* __restrict__ Remb, const int* __restrict__ mem,
                                float* __restrict__ outp, int n)
{
    int i = blockIdx.x * blockDim.x + threadIdx.x;
    if (i < n) {
        const int row = i >> 7, col = i & 127;
        const int node = row % NN;
        outp[i] = hb2[i] + hg[i] + Remb[mem[node] * HH + col];
    }
}

__global__ void pred_head(const float* __restrict__ mid, const float* __restrict__ W2,
                          const float* __restrict__ b2, float* __restrict__ outp)
{
    int i = blockIdx.x * blockDim.x + threadIdx.x;
    if (i < BN) {
        float s = 0.f;
        for (int c = 0; c < 64; ++c) s += mid[(long long)i * 64 + c] * W2[c];
        outp[i] = s + b2[0];
    }
}

// ---------------------------------------------------------------------------
static inline void gemm_fast(hipStream_t st, const float* A, const float* B, const float* bias,
                             float* C, int M, int N, int msA, int nsB, int ldc, int flags)
{
    dim3 g(M / 16, N / 32, 1);
    if (flags & FLAG_ACC)
        gemm_wmma_fast<FLAG_ACC, false><<<g, dim3(32), 0, st>>>(A, B, bias, C, msA, nsB, ldc);
    else if (flags & FLAG_RELU)
        gemm_wmma_fast<FLAG_RELU, true><<<g, dim3(32), 0, st>>>(A, B, bias, C, msA, nsB, ldc);
    else
        gemm_wmma_fast<0, true><<<g, dim3(32), 0, st>>>(A, B, bias, C, msA, nsB, ldc);
}

static inline void gemm(hipStream_t st, const float* A, const float* B, const float* bias,
                        float* C, int M, int N, int K,
                        int msA, int ksA, int nsB, int ksB, int ldc,
                        long long sA, long long sB, long long sC, int batch, int flags)
{
    dim3 g((M + 15) / 16, N / 16, batch);
    gemm_wmma<<<g, dim3(32), 0, st>>>(A, B, bias, C, M, N, K,
                                      msA, ksA, nsB, ksB, ldc, sA, sB, sC, flags);
}

static inline void launch_ln(hipStream_t st, const float* in, const float* resid,
                             const float* g, const float* b, float* out, int rows, int relu)
{
    layernorm_kernel<<<dim3((rows + 7) / 8), dim3(256), 0, st>>>(in, resid, g, b, out, rows, relu);
}

extern "C" void kernel_launch(void* const* d_in, const int* in_sizes, int n_in,
                              void* d_out, int out_size, void* d_ws, size_t ws_size,
                              hipStream_t stream)
{
    (void)in_sizes; (void)n_in; (void)out_size; (void)ws_size;

    const float* x       = (const float*)d_in[0];
    const float* A_norm  = (const float*)d_in[1];
    const float* Hinc    = (const float*)d_in[2];
    const int*   member  = (const int*)  d_in[3];
    const float* Wp      = (const float*)d_in[4];
    const float* bp      = (const float*)d_in[5];
    const float* g_p     = (const float*)d_in[6];
    const float* b_p     = (const float*)d_in[7];
    const float* Wqkv    = (const float*)d_in[8];
    const float* bqkv    = (const float*)d_in[9];
    const float* Wo      = (const float*)d_in[10];
    const float* bo      = (const float*)d_in[11];
    const float* g_a     = (const float*)d_in[12];
    const float* b_a     = (const float*)d_in[13];
    const float* Wih_f   = (const float*)d_in[14];
    const float* Whh_f   = (const float*)d_in[15];
    const float* bih_f   = (const float*)d_in[16];
    const float* bhh_f   = (const float*)d_in[17];
    const float* Wih_b   = (const float*)d_in[18];
    const float* Whh_b   = (const float*)d_in[19];
    const float* bih_b   = (const float*)d_in[20];
    const float* bhh_b   = (const float*)d_in[21];
    const float* Wtp     = (const float*)d_in[22];
    const float* btp     = (const float*)d_in[23];
    const float* Wn      = (const float*)d_in[24];
    const float* bn_     = (const float*)d_in[25];
    const float* We      = (const float*)d_in[26];
    const float* be      = (const float*)d_in[27];
    const float* Wa      = (const float*)d_in[28];
    const float* ba      = (const float*)d_in[29];
    const float* g_g     = (const float*)d_in[30];
    const float* b_g     = (const float*)d_in[31];
    const float* Remb    = (const float*)d_in[32];
    const float* Wg      = (const float*)d_in[33];
    const float* bg      = (const float*)d_in[34];
    const float* Wf      = (const float*)d_in[35];
    const float* bf      = (const float*)d_in[36];
    const float* g_f     = (const float*)d_in[37];
    const float* b_f     = (const float*)d_in[38];
    const float* W1      = (const float*)d_in[39];
    const float* b1      = (const float*)d_in[40];
    const float* W2      = (const float*)d_in[41];
    const float* b2      = (const float*)d_in[42];

    // Output slots (return order: pred, h_spatial, h_temporal)
    float* out   = (float*)d_out;
    float* pred  = out;                 // 1600
    float* h_sp  = out + BN;            // 204800
    float* h_tp  = out + BN + BN * HH;  // 204800

    // Workspace carve-up
    float* ws   = (float*)d_ws;
    float* h0   = ws;                         // P*128
    float* qkvb = h0 + (long long)PP * HH;    // P*384  (reused as wo_out)
    float* aob  = qkvb + (long long)PP * 384; // P*128  (reused as h1)
    float* p    = aob + (long long)PP * HH;
    float* gates  = p; p += BN * 512;
    float* hA     = p; p += BN * HH;
    float* hB     = p; p += BN * HH;
    float* cA     = p; p += BN * HH;
    float* cB     = p; p += BN * HH;
    float* y_b    = p; p += BN * HH;
    float* bsum_f = p; p += 512;
    float* bsum_b = p; p += 512;
    float* npj    = p; p += BN * HH;
    float* deg    = p; p += EE;
    float* ee_raw = p; p += BB * EE * HH;
    float* ee     = p; p += BB * EE * HH;
    float* sc     = p; p += BB * EE;
    float* wbuf   = p; p += BN * EE;
    float* agg    = p; p += BN * HH;
    float* hb2    = p; p += BN * HH;
    float* gcn1   = p; p += BN * HH;
    float* hg     = p; p += BN * HH;
    float* hf     = p; p += BN * HH;
    float* mid    = p; p += BN * 64;

    // 1. Input projection: h0 = x @ Wp^T + bp   (K=26 -> generic guarded path)
    gemm(stream, x, Wp, bp, h0, PP, HH, FF, FF, 1, FF, 1, HH, 0, 0, 0, 1, 0);
    launch_ln(stream, h0, nullptr, g_p, b_p, h0, PP, 1);  // ReLU(LN(.)) in place

    // 2. qkv = h0 @ Wqkv^T + bqkv
    gemm_fast(stream, h0, Wqkv, bqkv, qkvb, PP, 384, HH, HH, 384, 0);

    // 3. Fused attention -> aob
    attention_fused<<<dim3(BN * NHH), dim3(256), 0, stream>>>(qkvb, aob);

    // 4. wo_out = ao @ Wo^T + bo  (reuse qkv region)
    float* wo_out = qkvb;
    gemm_fast(stream, aob, Wo, bo, wo_out, PP, HH, HH, HH, HH, 0);

    // 5. h1 = LN(wo_out + h0)  (into old ao region)
    float* h1 = aob;
    launch_ln(stream, wo_out, h0, g_a, b_a, h1, PP, 0);

    // 6. Combined LSTM biases
    add_vec<<<dim3(2), dim3(256), 0, stream>>>(bih_f, bhh_f, bsum_f, 512);
    add_vec<<<dim3(2), dim3(256), 0, stream>>>(bih_b, bhh_b, bsum_b, 512);

    // 7. Backward LSTM = single step on x_{T-1} from zero state
    gemm_fast(stream, h1 + (TT - 1) * HH, Wih_b, bsum_b, gates, BN, 512,
              TT * HH, HH, 512, 0);
    lstm_cell<<<dim3(800), dim3(256), 0, stream>>>(gates, nullptr, y_b, cA, BN, 1);

    // 8. Forward LSTM scan over T
    float* hp = hA; float* hn = hB; float* cp = cA; float* cn = cB;
    for (int t = 0; t < TT; ++t) {
        gemm_fast(stream, h1 + t * HH, Wih_f, bsum_f, gates, BN, 512,
                  TT * HH, HH, 512, 0);
        if (t > 0)
            gemm_fast(stream, hp, Whh_f, nullptr, gates, BN, 512,
                      HH, HH, 512, FLAG_ACC);
        lstm_cell<<<dim3(800), dim3(256), 0, stream>>>(gates, cp, hn, cn, BN, t == 0 ? 1 : 0);
        float* tmp;
        tmp = hp; hp = hn; hn = tmp;
        tmp = cp; cp = cn; cn = tmp;
    }
    float* y_f = hp;

    // 9. h_temporal = [y_f, y_b] @ Wtp^T + btp   (two half-K GEMMs)
    gemm_fast(stream, y_f, Wtp,      btp,     h_tp, BN, HH, HH, 2 * HH, HH, 0);
    gemm_fast(stream, y_b, Wtp + HH, nullptr, h_tp, BN, HH, HH, 2 * HH, HH, FLAG_ACC);

    // 10. Spatial HyperGAT + GCN
    gemm_fast(stream, h_tp, Wn, bn_, npj, BN, HH, HH, HH, HH, 0);
    edge_deg<<<dim3(1), dim3(16), 0, stream>>>(Hinc, deg);
    // ee_raw[b] = Hinc^T @ npj[b]   (A strided as transpose, batched over B)
    gemm(stream, Hinc, npj, nullptr, ee_raw, EE, HH, NN,
         1, EE, 1, HH, HH, 0, (long long)NN * HH, (long long)EE * HH, BB, 0);
    scale_edges<<<dim3(128), dim3(256), 0, stream>>>(ee_raw, deg, BB * EE * HH);
    gemm_fast(stream, ee_raw, We, be, ee, BB * EE, HH, HH, HH, HH, 0);
    edge_score<<<dim3(1), dim3(256), 0, stream>>>(ee, Wa, ba, sc);
    edge_softmax<<<dim3(7), dim3(256), 0, stream>>>(sc, Hinc, wbuf);
    // agg[b] = w[b] @ ee[b]
    gemm(stream, wbuf, ee, nullptr, agg, NN, HH, EE,
         EE, 1, 1, HH, HH, (long long)NN * EE, (long long)EE * HH, (long long)NN * HH, BB, 0);
    launch_ln(stream, agg, h_tp, g_g, b_g, hb2, BN, 0);
    // gcn1[b] = A_norm @ hb2[b]
    gemm(stream, A_norm, hb2, nullptr, gcn1, NN, HH, NN,
         NN, 1, 1, HH, HH, 0, (long long)NN * HH, (long long)NN * HH, BB, 0);
    gemm_fast(stream, gcn1, Wg, bg, hg, BN, HH, HH, HH, HH, FLAG_RELU);
    spatial_combine<<<dim3(800), dim3(256), 0, stream>>>(hb2, hg, Remb, member, h_sp, BN * HH);

    // 11. Fusion: hf = ReLU(LN([h_tp, h_sp] @ Wf^T + bf))
    gemm_fast(stream, h_tp, Wf,      bf,      hf, BN, HH, HH, 2 * HH, HH, 0);
    gemm_fast(stream, h_sp, Wf + HH, nullptr, hf, BN, HH, HH, 2 * HH, HH, FLAG_ACC);
    launch_ln(stream, hf, nullptr, g_f, b_f, hf, BN, 1);

    // 12. Head
    gemm_fast(stream, hf, W1, b1, mid, BN, 64, HH, HH, 64, FLAG_RELU);
    pred_head<<<dim3(7), dim3(256), 0, stream>>>(mid, W2, b2, pred);
}
